// InteractionLayer_40003325395258
// MI455X (gfx1250) — matrix operand order repaired and checked by reading, not compiled, not run on hardware
//
#include <hip/hip_runtime.h>
#include <math.h>

#define D   128
#define D3  384
#define PAD_BF 136   // bf16 row stride: 272B -> bank-group step 4, conflict-free b128 reads
#define PAD_W  388   // f32 W-tile row stride

typedef __attribute__((ext_vector_type(16))) __bf16 v16bf;
typedef __attribute__((ext_vector_type(8)))  float  v8f;
typedef __attribute__((ext_vector_type(4)))  unsigned int v4u;
typedef __attribute__((ext_vector_type(8)))  int v8i;
typedef __attribute__((ext_vector_type(4)))  int v4i;

__device__ __forceinline__ float sigmoidf_(float x) { return 1.0f / (1.0f + __expf(-x)); }

// ---- bf16 split helpers (round-to-nearest-even hi, exact residual lo) ------
__device__ __forceinline__ unsigned short bfbits(float x) {
    unsigned u = __float_as_uint(x);
    return (unsigned short)((u + 0x7FFFu + ((u >> 16) & 1u)) >> 16);
}
__device__ __forceinline__ float bffloat(unsigned short s) {
    return __uint_as_float(((unsigned)s) << 16);
}

__device__ __forceinline__ v16bf mkfrag(uint4 a, uint4 b) {
    union { uint4 q[2]; v16bf v; } u;
    u.q[0] = a; u.q[1] = b;
    return u.v;
}
// A fragment (16x32 bf16): VGPR0-3 = K(kbase+16*half .. +7), VGPR4-7 = K(kbase+16*half+8 .. +15)
__device__ __forceinline__ v16bf loadA(const unsigned short* rowk, int half) {
    const uint4* p0 = (const uint4*)(rowk + 16 * half);
    const uint4* p1 = (const uint4*)(rowk + 16 * half + 8);
    return mkfrag(*p0, *p1);
}
#define WMMA_BF16(A, B, C) \
    __builtin_amdgcn_wmma_f32_16x16x32_bf16(false, (A), false, (B), (short)0, (C), false, false)

// ---------------------------------------------------------------------------
// Tensor Data Mover: DMA a 64-row x 128-col f32 tile (row-major, stride 128)
// from global memory into LDS at byte offset lds_off. rows_avail rows are
// valid; TDM zero-fills reads past tensor_dim1 (handles the tail tile).
// D# layout per CDNA5 ISA ch.8 (group0 128b, group1 256b; groups 2/3 zero).
// ---------------------------------------------------------------------------
__device__ __forceinline__ void tdm_load_tile_f32(const float* gsrc, unsigned lds_off,
                                                  unsigned rows_avail) {
    unsigned long long ga = (unsigned long long)(const void*)gsrc;
    v4u g0;
    g0[0] = 1u;                                            // count=1, user mode
    g0[1] = lds_off;                                       // LDS byte address
    g0[2] = (unsigned)(ga & 0xFFFFFFFFull);                // global_addr[31:0]
    g0[3] = (unsigned)((ga >> 32) & 0x1FFFFFFull)          // global_addr[56:32]
          | (2u << 30);                                    // type=2 ("image")
    v8i g1;
    g1[0] = (int)(2u << 16);                               // wg_mask=0, data_size=4B
    g1[1] = (int)(128u << 16);                             // tensor_dim0 = 128
    g1[2] = (int)((rows_avail & 0xFFFFu) << 16);           // tensor_dim1[15:0]
    g1[3] = (int)(((rows_avail >> 16) & 0xFFFFu)           // tensor_dim1[31:16]
          | (128u << 16));                                 // tile_dim0 = 128
    g1[4] = (int)64u;                                      // tile_dim1 = 64, tile_dim2 = 0
    g1[5] = (int)128u;                                     // tensor_dim0_stride = 128
    g1[6] = 0;
    g1[7] = 0;
    v4i gz = {0, 0, 0, 0};
#if defined(__clang_major__) && (__clang_major__ >= 23)
    v8i gz8 = {0, 0, 0, 0, 0, 0, 0, 0};
    __builtin_amdgcn_tensor_load_to_lds(g0, g1, gz, gz, gz8, 0);
#else
    __builtin_amdgcn_tensor_load_to_lds(g0, g1, gz, gz, 0);
#endif
}

// ---------------------------------------------------------------------------
// Kernel 0: split weights into bf16 hi/lo planes (run once per call; tiny)
// ---------------------------------------------------------------------------
__global__ void wconv_kernel(const float* __restrict__ w,
                             unsigned short* __restrict__ hi,
                             unsigned short* __restrict__ lo, int n) {
    int i = blockIdx.x * blockDim.x + threadIdx.x;
    if (i < n) {
        float x = w[i];
        unsigned short h = bfbits(x);
        hi[i] = h;
        lo[i] = bfbits(x - bffloat(h));
    }
}

// ---------------------------------------------------------------------------
// Kernel 1: graph (group) layer norm -> out_s (also the "s +" scatter base)
// ---------------------------------------------------------------------------
__global__ __launch_bounds__(256)
void ln_kernel(const float* __restrict__ s, const int* __restrict__ splits,
               const float* __restrict__ gamma, const float* __restrict__ beta,
               float* __restrict__ out_s) {
    __shared__ float red_s[256], red_q[256];
    __shared__ float sh_mean, sh_rstd;
    const int g = blockIdx.x;
    int start = 0;
    for (int i = 0; i < g; ++i) start += splits[i];
    const int  cnt   = splits[g];
    const long base  = (long)start * D;
    const long total = (long)cnt * D;

    float sum = 0.f, sq = 0.f;
    for (long i = threadIdx.x; i < total; i += blockDim.x) {
        float x = s[base + i];
        sum += x; sq += x * x;
    }
    red_s[threadIdx.x] = sum; red_q[threadIdx.x] = sq;
    __syncthreads();
    for (int off = 128; off > 0; off >>= 1) {
        if (threadIdx.x < off) {
            red_s[threadIdx.x] += red_s[threadIdx.x + off];
            red_q[threadIdx.x] += red_q[threadIdx.x + off];
        }
        __syncthreads();
    }
    if (threadIdx.x == 0) {
        float inv  = 1.0f / (float)total;
        float mean = red_s[0] * inv;
        float var  = red_q[0] * inv - mean * mean;
        sh_mean = mean;
        sh_rstd = rsqrtf(var + 1e-5f);
    }
    __syncthreads();
    const float mean = sh_mean, rstd = sh_rstd;
    for (long i = threadIdx.x; i < total; i += blockDim.x) {
        int d = (int)(i & (D - 1));
        float x = s[base + i];
        out_s[base + i] = (x - mean) * rstd * gamma[d] + beta[d];
    }
}

// ---------------------------------------------------------------------------
// Kernel 2: out_v = node_states_v (float4 grid-stride copy)
// ---------------------------------------------------------------------------
__global__ void copy_v_kernel(const float4* __restrict__ src, float4* __restrict__ dst, long n4) {
    long i = (long)blockIdx.x * blockDim.x + threadIdx.x;
    long stride = (long)gridDim.x * blockDim.x;
    for (; i < n4; i += stride) dst[i] = src[i];
}

// ---------------------------------------------------------------------------
// Kernel 3: phi = (silu(s_norm @ m1^T + b1)) @ m2^T + b2
// TDM tile staging + bf16x3 split-precision WMMA; h kept as bf16 hi/lo in LDS.
// ---------------------------------------------------------------------------
__global__ __launch_bounds__(256)
void node_phi_kernel(const float* __restrict__ s_norm,
                     const unsigned short* __restrict__ w1h, const unsigned short* __restrict__ w1l,
                     const float* __restrict__ b1,
                     const unsigned short* __restrict__ w2h, const unsigned short* __restrict__ w2l,
                     const float* __restrict__ b2,
                     float* __restrict__ phi, int N_) {
    __shared__ __align__(16) unsigned char smem[4 * 64 * PAD_BF * 2];   // 69632 B
    unsigned short* sAh = (unsigned short*)smem;
    unsigned short* sAl = sAh + 64 * PAD_BF;
    unsigned short* sHh = sAl + 64 * PAD_BF;
    unsigned short* sHl = sHh + 64 * PAD_BF;
    float* sF32 = (float*)sHh;   // 32KB f32 staging aliases sH planes (staging-only)

    const int tid  = threadIdx.x;
    const int wave = tid >> 5;
    const int lane = tid & 31;
    const int r    = lane & 15;
    const int half = lane >> 4;
    const int row0 = blockIdx.x * 64;

    // TDM: DMA the raw 64x128 f32 s tile into LDS (zero-fill past N)
    if (wave == 0) {
        tdm_load_tile_f32(s_norm + (long)row0 * D, (unsigned)(size_t)(void*)sF32,
                          (unsigned)(N_ - row0));
        __builtin_amdgcn_s_wait_tensorcnt(0);
    }
    __syncthreads();

    // split f32 tile -> bf16 hi/lo planes
    for (int i = tid; i < 64 * 32; i += 256) {
        int rr = i >> 5, cc = (i & 31) << 2;
        float4 v = *(const float4*)(sF32 + rr * D + cc);
        unsigned short h0 = bfbits(v.x), h1 = bfbits(v.y), h2 = bfbits(v.z), h3 = bfbits(v.w);
        unsigned short l0 = bfbits(v.x - bffloat(h0)), l1 = bfbits(v.y - bffloat(h1));
        unsigned short l2 = bfbits(v.z - bffloat(h2)), l3 = bfbits(v.w - bffloat(h3));
        uint2 ph = make_uint2(((unsigned)h1 << 16) | h0, ((unsigned)h3 << 16) | h2);
        uint2 pl = make_uint2(((unsigned)l1 << 16) | l0, ((unsigned)l3 << 16) | l2);
        *(uint2*)&sAh[rr * PAD_BF + cc] = ph;
        *(uint2*)&sAl[rr * PAD_BF + cc] = pl;
    }
    __syncthreads();

    // phase 1: h = silu(sA @ w1^T + b1) -> sH planes  (8 col-tiles, 1 per wave)
    for (int nt = wave; nt < 8; nt += 8) {
        uint4 bh0[4], bh1[4], bl0[4], bl1[4];
        const unsigned short* brh = w1h + (nt * 16 + r) * D;
        const unsigned short* brl = w1l + (nt * 16 + r) * D;
        #pragma unroll
        for (int ks = 0; ks < 4; ++ks) {
            const uint4* p = (const uint4*)(brh + ks * 32 + 16 * half);
            bh0[ks] = p[0]; bh1[ks] = p[1];
            const uint4* q = (const uint4*)(brl + ks * 32 + 16 * half);
            bl0[ks] = q[0]; bl1[ks] = q[1];
        }
        int col = nt * 16 + r;
        float bias = b1[col];
        for (int mt = 0; mt < 4; ++mt) {
            v8f acc = {};
            const unsigned short* arh = sAh + (mt * 16 + r) * PAD_BF;
            const unsigned short* arl = sAl + (mt * 16 + r) * PAD_BF;
            #pragma unroll
            for (int ks = 0; ks < 4; ++ks) {
                v16bf ah = loadA(arh + ks * 32, half);
                v16bf al = loadA(arl + ks * 32, half);
                v16bf bh = mkfrag(bh0[ks], bh1[ks]);
                v16bf bl = mkfrag(bl0[ks], bl1[ks]);
                acc = WMMA_BF16(ah, bh, acc);
                acc = WMMA_BF16(ah, bl, acc);
                acc = WMMA_BF16(al, bh, acc);
            }
            #pragma unroll
            for (int v = 0; v < 8; ++v) {
                int row = mt * 16 + v + 8 * half;
                float x = acc[v] + bias;
                float hsl = x * sigmoidf_(x);            // silu
                unsigned short hb = bfbits(hsl);
                sHh[row * PAD_BF + col] = hb;
                sHl[row * PAD_BF + col] = bfbits(hsl - bffloat(hb));
            }
        }
    }
    __syncthreads();

    // phase 2: phi = sH @ w2^T + b2  (24 col-tiles, 3 per wave; B frags cached)
    for (int nt = wave; nt < 24; nt += 8) {
        uint4 bh0[4], bh1[4], bl0[4], bl1[4];
        const unsigned short* brh = w2h + (nt * 16 + r) * D;
        const unsigned short* brl = w2l + (nt * 16 + r) * D;
        #pragma unroll
        for (int ks = 0; ks < 4; ++ks) {
            const uint4* p = (const uint4*)(brh + ks * 32 + 16 * half);
            bh0[ks] = p[0]; bh1[ks] = p[1];
            const uint4* q = (const uint4*)(brl + ks * 32 + 16 * half);
            bl0[ks] = q[0]; bl1[ks] = q[1];
        }
        int col = nt * 16 + r;
        float bias = b2[col];
        for (int mt = 0; mt < 4; ++mt) {
            v8f acc = {};
            const unsigned short* arh = sHh + (mt * 16 + r) * PAD_BF;
            const unsigned short* arl = sHl + (mt * 16 + r) * PAD_BF;
            #pragma unroll
            for (int ks = 0; ks < 4; ++ks) {
                v16bf ah = loadA(arh + ks * 32, half);
                v16bf al = loadA(arl + ks * 32, half);
                v16bf bh = mkfrag(bh0[ks], bh1[ks]);
                v16bf bl = mkfrag(bl0[ks], bl1[ks]);
                acc = WMMA_BF16(ah, bh, acc);
                acc = WMMA_BF16(ah, bl, acc);
                acc = WMMA_BF16(al, bh, acc);
            }
            #pragma unroll
            for (int v = 0; v < 8; ++v) {
                int row = row0 + mt * 16 + v + 8 * half;
                if (row < N_) phi[(long)row * D3 + col] = acc[v] + bias;
            }
        }
    }
}

// ---------------------------------------------------------------------------
// Kernel 4: fused edge pipeline. 64-edge tiles:
//   TDM-stage edge_states tile; W = tile @ W_w^T + W_b (bf16x3 WMMA -> LDS)
//   Wphi = W * phi[src];  g = sigmoid(Wphi_s . edge_w + edge_b)
//   atomic scatter of msg_s / msg_v into out_s / out_v by dst.
// ---------------------------------------------------------------------------
__global__ __launch_bounds__(256)
void edge_kernel(const float* __restrict__ edge_states,
                 const unsigned short* __restrict__ Wwh, const unsigned short* __restrict__ Wwl,
                 const float* __restrict__ Wb,
                 const float* __restrict__ phi,
                 const int*   __restrict__ edges,
                 const float* __restrict__ node_v,
                 const float* __restrict__ uv,
                 const float* __restrict__ ew, const float* __restrict__ eb,
                 float* __restrict__ out_s, float* __restrict__ out_v, int E_) {
    __shared__ __align__(16) unsigned char smem[2 * 64 * PAD_BF * 2 + 64 * PAD_W * 4]; // 134144 B
    unsigned short* sAh = (unsigned short*)smem;
    unsigned short* sAl = sAh + 64 * PAD_BF;
    float* sW   = (float*)(smem + 2 * 64 * PAD_BF * 2);
    float* sF32 = sW;            // 32KB f32 staging aliases the W tile (staging-only)

    const int tid  = threadIdx.x;
    const int wave = tid >> 5;
    const int lane = tid & 31;
    const int r    = lane & 15;
    const int half = lane >> 4;
    const int e0   = blockIdx.x * 64;

    // TDM: DMA the raw 64x128 f32 edge_states tile into LDS (zero-fill past E)
    if (wave == 0) {
        tdm_load_tile_f32(edge_states + (long)e0 * D, (unsigned)(size_t)(void*)sF32,
                          (unsigned)(E_ - e0));
        __builtin_amdgcn_s_wait_tensorcnt(0);
    }
    __syncthreads();

    for (int i = tid; i < 64 * 32; i += 256) {
        int rr = i >> 5, cc = (i & 31) << 2;
        float4 v = *(const float4*)(sF32 + rr * D + cc);
        unsigned short h0 = bfbits(v.x), h1 = bfbits(v.y), h2 = bfbits(v.z), h3 = bfbits(v.w);
        unsigned short l0 = bfbits(v.x - bffloat(h0)), l1 = bfbits(v.y - bffloat(h1));
        unsigned short l2 = bfbits(v.z - bffloat(h2)), l3 = bfbits(v.w - bffloat(h3));
        uint2 ph = make_uint2(((unsigned)h1 << 16) | h0, ((unsigned)h3 << 16) | h2);
        uint2 pl = make_uint2(((unsigned)l1 << 16) | l0, ((unsigned)l3 << 16) | l2);
        *(uint2*)&sAh[rr * PAD_BF + cc] = ph;
        *(uint2*)&sAl[rr * PAD_BF + cc] = pl;
    }
    __syncthreads();

    // W tile: 24 col-tiles, 3 per wave, B frags cached over 4 row-tiles
    for (int nt = wave; nt < 24; nt += 8) {
        uint4 bh0[4], bh1[4], bl0[4], bl1[4];
        const unsigned short* brh = Wwh + (nt * 16 + r) * D;
        const unsigned short* brl = Wwl + (nt * 16 + r) * D;
        #pragma unroll
        for (int ks = 0; ks < 4; ++ks) {
            const uint4* p = (const uint4*)(brh + ks * 32 + 16 * half);
            bh0[ks] = p[0]; bh1[ks] = p[1];
            const uint4* q = (const uint4*)(brl + ks * 32 + 16 * half);
            bl0[ks] = q[0]; bl1[ks] = q[1];
        }
        int col = nt * 16 + r;
        float bias = Wb[col];
        for (int mt = 0; mt < 4; ++mt) {
            v8f acc = {};
            const unsigned short* arh = sAh + (mt * 16 + r) * PAD_BF;
            const unsigned short* arl = sAl + (mt * 16 + r) * PAD_BF;
            #pragma unroll
            for (int ks = 0; ks < 4; ++ks) {
                v16bf ah = loadA(arh + ks * 32, half);
                v16bf al = loadA(arl + ks * 32, half);
                v16bf bh = mkfrag(bh0[ks], bh1[ks]);
                v16bf bl = mkfrag(bl0[ks], bl1[ks]);
                acc = WMMA_BF16(ah, bh, acc);
                acc = WMMA_BF16(ah, bl, acc);
                acc = WMMA_BF16(al, bh, acc);
            }
            #pragma unroll
            for (int v = 0; v < 8; ++v) {
                int row = mt * 16 + v + 8 * half;
                sW[row * PAD_W + col] = acc[v] + bias;
            }
        }
    }
    __syncthreads();

    // epilogue: each wave handles 8 edges; 32 lanes cover D=128, 4 chans each
    const float eb0 = eb[0];
    for (int le = wave * 8; le < wave * 8 + 8; ++le) {
        int e = e0 + le;
        if (e >= E_) break;
        int src = edges[2 * e + 0];
        int dst = edges[2 * e + 1];
        const float* ph  = phi + (long)src * D3;
        const float* swr = sW + le * PAD_W;

        float wps[4];
        float dot = 0.f;
        #pragma unroll
        for (int j = 0; j < 4; ++j) {
            int c = lane + 32 * j;
            wps[j] = swr[c] * ph[c];       // Wphi_s
            dot   += wps[j] * ew[c];
        }
        #pragma unroll
        for (int m = 16; m >= 1; m >>= 1) dot += __shfl_xor(dot, m, 32);
        const float g = sigmoidf_(dot + eb0);

        const float u0 = uv[3 * e + 0], u1 = uv[3 * e + 1], u2 = uv[3 * e + 2];
        const float* nv = node_v + (long)src * D3;
        float* os = out_s + (long)dst * D;
        float* ov = out_v + (long)dst * D3;
        #pragma unroll
        for (int j = 0; j < 4; ++j) {
            int c = lane + 32 * j;
            atomicAdd(os + c, wps[j] * g);
            float wvv = swr[D   + c] * ph[D   + c];   // Wphi_vv
            float wvs = swr[2*D + c] * ph[2*D + c];   // Wphi_vs
            atomicAdd(ov + c,        (nv[c]       * wvv + wvs * u0) * g);
            atomicAdd(ov + D + c,    (nv[D + c]   * wvv + wvs * u1) * g);
            atomicAdd(ov + 2*D + c,  (nv[2*D + c] * wvv + wvs * u2) * g);
        }
    }
}

// ---------------------------------------------------------------------------
extern "C" void kernel_launch(void* const* d_in, const int* in_sizes, int n_in,
                              void* d_out, int out_size, void* d_ws, size_t ws_size,
                              hipStream_t stream) {
    const float* node_s      = (const float*)d_in[0];
    const float* node_v      = (const float*)d_in[1];
    const float* edge_states = (const float*)d_in[2];
    const float* unit_vec    = (const float*)d_in[3];
    const int*   edges       = (const int*)  d_in[4];
    const int*   splits      = (const int*)  d_in[5];
    const float* W_w  = (const float*)d_in[6];
    const float* W_b  = (const float*)d_in[7];
    const float* m1w  = (const float*)d_in[8];
    const float* m1b  = (const float*)d_in[9];
    const float* m2w  = (const float*)d_in[10];
    const float* m2b  = (const float*)d_in[11];
    const float* e_w  = (const float*)d_in[12];
    const float* e_b  = (const float*)d_in[13];
    const float* gam  = (const float*)d_in[14];
    const float* bet  = (const float*)d_in[15];

    const int N_ = in_sizes[0] / D;
    const int E_ = in_sizes[2] / D;
    const int G_ = in_sizes[5];

    float* out_s = (float*)d_out;
    float* out_v = out_s + (long)N_ * D;

    // workspace layout: phi (N x 384 f32), then bf16 hi/lo weight planes
    char*  ws    = (char*)d_ws;
    float* phi   = (float*)ws;
    size_t off   = (((size_t)N_ * D3 * 4) + 255) & ~(size_t)255;
    unsigned short* w1h = (unsigned short*)(ws + off); off += (size_t)D  * D * 2;
    unsigned short* w1l = (unsigned short*)(ws + off); off += (size_t)D  * D * 2;
    unsigned short* w2h = (unsigned short*)(ws + off); off += (size_t)D3 * D * 2;
    unsigned short* w2l = (unsigned short*)(ws + off); off += (size_t)D3 * D * 2;
    unsigned short* wwh = (unsigned short*)(ws + off); off += (size_t)D3 * D * 2;
    unsigned short* wwl = (unsigned short*)(ws + off);

    // 0. split weights into bf16 hi/lo planes
    wconv_kernel<<<(D * D  + 255) / 256, 256, 0, stream>>>(m1w, w1h, w1l, D * D);
    wconv_kernel<<<(D3 * D + 255) / 256, 256, 0, stream>>>(m2w, w2h, w2l, D3 * D);
    wconv_kernel<<<(D3 * D + 255) / 256, 256, 0, stream>>>(W_w, wwh, wwl, D3 * D);

    // 1. group layer norm -> out_s
    ln_kernel<<<G_, 256, 0, stream>>>(node_s, splits, gam, bet, out_s);

    // 2. out_v = node_states_v
    long n4 = (long)N_ * D3 / 4;
    copy_v_kernel<<<1024, 256, 0, stream>>>((const float4*)node_v, (float4*)out_v, n4);

    // 3. phi = (silu(s_norm @ m1^T + b1)) @ m2^T + b2
    node_phi_kernel<<<(N_ + 63) / 64, 256, 0, stream>>>(out_s, w1h, w1l, m1b,
                                                        w2h, w2l, m2b, phi, N_);

    // 4. fused edge GEMM + gate + atomic scatter
    edge_kernel<<<(E_ + 63) / 64, 256, 0, stream>>>(edge_states, wwh, wwl, W_b, phi, edges,
                                                    node_v, unit_vec, e_w, e_b,
                                                    out_s, out_v, E_);
}
